// OnlineTripletLoss_1082331758628
// MI455X (gfx1250) — matrix operand order repaired and checked by reading, not compiled, not run on hardware
//
#include <hip/hip_runtime.h>
#include <math.h>

#define E 1024
#define MARGIN 1.0f
#define ROWS_PB 128   // rows per block (8 waves * 16)
#define COLS_PI 128   // column group per iteration (8 subtiles of 16)
#define KC 32         // K chunk (bf16 WMMA K)
#define NK (E / KC)
#define NSPLIT 8      // column splits across grid.y

// Use inline-asm CDNA5 path: untransposed LDS staging + ds_load_b128
// direct reads. Set to 0 to fall back to the proven compiler-only path.
#define USE_ASM_TR16 1

#if defined(__has_builtin)
#  if __has_builtin(__builtin_amdgcn_tensor_load_to_lds) && __has_builtin(__builtin_amdgcn_s_wait_tensorcnt)
#    define HAVE_TDM 1
#  endif
#endif
#ifndef HAVE_TDM
#  define HAVE_TDM 0
#endif

typedef __attribute__((ext_vector_type(16))) __bf16 v16bf;
typedef __attribute__((ext_vector_type(8)))  float  v8f;
typedef __attribute__((ext_vector_type(4)))  float  f4;
typedef __attribute__((ext_vector_type(2)))  unsigned int u2;
typedef __attribute__((ext_vector_type(4)))  unsigned int ui4;
typedef __attribute__((ext_vector_type(8)))  int    i8v;
typedef __attribute__((ext_vector_type(4)))  int    i4v;

__device__ __forceinline__ unsigned short f2bf(float f) {
  unsigned u = __float_as_uint(f);
  u += 0x7FFFu + ((u >> 16) & 1u);   // round-to-nearest-even
  return (unsigned short)(u >> 16);
}
// order-preserving float -> u32 map (monotone), and inverse
__device__ __forceinline__ unsigned orderF(float f) {
  unsigned u = __float_as_uint(f);
  return (u & 0x80000000u) ? ~u : (u | 0x80000000u);
}
__device__ __forceinline__ float unorderF(unsigned v) {
  unsigned u = (v & 0x80000000u) ? (v & 0x7FFFFFFFu) : ~v;
  return __uint_as_float(u);
}

// ---------------------------------------------------------------------------
// Kernel 1: per-row norms + cosine(anchor,positive), write bf16 a_n / p_n
// ---------------------------------------------------------------------------
__global__ __launch_bounds__(256)
void normalize_k(const float* __restrict__ anchor,
                 const float* __restrict__ positive,
                 unsigned short* __restrict__ a16,
                 unsigned short* __restrict__ p16,
                 float* __restrict__ ap) {
  const int row = blockIdx.x;
  const int tid = threadIdx.x;           // 256 threads, 4 floats each
  const f4 a = ((const f4*)(anchor   + (size_t)row * E))[tid];
  const f4 p = ((const f4*)(positive + (size_t)row * E))[tid];
  float sa = a.x*a.x + a.y*a.y + a.z*a.z + a.w*a.w;
  float sp = p.x*p.x + p.y*p.y + p.z*p.z + p.w*p.w;
  float sd = a.x*p.x + a.y*p.y + a.z*p.z + a.w*p.w;

  __shared__ float rA[256], rP[256], rD[256];
  __shared__ float inv[2];
  rA[tid] = sa; rP[tid] = sp; rD[tid] = sd;
  __syncthreads();
  for (int off = 128; off > 0; off >>= 1) {
    if (tid < off) { rA[tid] += rA[tid+off]; rP[tid] += rP[tid+off]; rD[tid] += rD[tid+off]; }
    __syncthreads();
  }
  if (tid == 0) {
    float na = sqrtf(rA[0]), np = sqrtf(rP[0]);
    inv[0] = 1.0f / na; inv[1] = 1.0f / np;
    ap[row] = rD[0] / fmaxf(na * np, 1e-8f);
  }
  __syncthreads();
  const float ia = inv[0], ip = inv[1];
  u2 oa, op;
  oa.x = (unsigned)f2bf(a.x*ia) | ((unsigned)f2bf(a.y*ia) << 16);
  oa.y = (unsigned)f2bf(a.z*ia) | ((unsigned)f2bf(a.w*ia) << 16);
  op.x = (unsigned)f2bf(p.x*ip) | ((unsigned)f2bf(p.y*ip) << 16);
  op.y = (unsigned)f2bf(p.z*ip) | ((unsigned)f2bf(p.w*ip) << 16);
  ((u2*)(a16 + (size_t)row * E))[tid] = oa;
  ((u2*)(p16 + (size_t)row * E))[tid] = op;
}

// ---------------------------------------------------------------------------
// Kernel 2: init packed (d,sim) argmax accumulators
// ---------------------------------------------------------------------------
__global__ void init_k(unsigned long long* __restrict__ packed, int n) {
  int i = blockIdx.x * blockDim.x + threadIdx.x;
  if (i < n) packed[i] = 0ull;
}

// ---------------------------------------------------------------------------
// Kernel 3: fused bf16 WMMA GEMM (sim = a_n @ p_n^T) + hardest-negative mining
// TDM (tensor_load_to_lds) stages 128x32 bf16 tiles into double-buffered LDS;
// ds_load_b128 direct reads build B fragments; counters managed
// manually (TENSORcnt for staging, DScnt exclusively for fragment reads).
// ---------------------------------------------------------------------------
__global__ __launch_bounds__(256)
void mine_k(const unsigned short* __restrict__ a16,
            const unsigned short* __restrict__ p16,
            unsigned long long* __restrict__ packed,
            int B) {
#if USE_ASM_TR16
  __shared__ unsigned short Bt[2][COLS_PI][KC];   // 16 KB, row-major (N x K)
#else
  __shared__ unsigned short Bt[2][KC][COLS_PI];   // 16 KB, transposed (K x N)
#endif
  const int tid    = threadIdx.x;
  const int lane   = tid & 31;
  const int wave   = tid >> 5;
  const int lane15 = lane & 15;
  const int hsel   = lane >> 4;                 // lane half selects A K-subgroup
  const int rbase  = blockIdx.x * ROWS_PB + wave * 16;
  const int cps    = B / NSPLIT;
  const int cbeg   = blockIdx.y * cps;

  float bestD[8], bestS[8];
#pragma unroll
  for (int r = 0; r < 8; ++r) { bestD[r] = -INFINITY; bestS[r] = 0.0f; }

  const unsigned short* arow = a16 + (size_t)(rbase + lane15) * E;

  // stage K-chunk [kk, kk+32) of p rows [jj, jj+128) into LDS buffer bufi
  auto stage = [&](int bufi, int jj, int kk) {
#if USE_ASM_TR16 && HAVE_TDM
    if (wave == 0) {   // one TDM descriptor per chunk, tracked by TENSORcnt
      unsigned long long ga = (unsigned long long)(size_t)(p16 + (size_t)jj * E + kk);
      unsigned ldsoff = (unsigned)(size_t)&Bt[bufi][0][0];
      ui4 g0;
      g0.x = 1u;                                     // count=1, user descriptor
      g0.y = ldsoff;                                 // lds_addr
      g0.z = (unsigned)ga;                           // global_addr[31:0]
      g0.w = (unsigned)(ga >> 32) | (2u << 30);      // global_addr[56:32] | type=2
      i8v g1 = {};
      g1[0] = 1 << 16;                               // data_size = 2 bytes
      g1[1] = (int)((unsigned)(E & 0xFFFF) << 16);   // tensor_dim0[15:0]
      g1[2] = (int)(((unsigned)B & 0xFFFFu) << 16) | ((E >> 16) & 0xFFFF); // dim0 hi | dim1 lo
      g1[3] = (int)((KC & 0xFFFF) << 16) | (((unsigned)B >> 16) & 0xFFFF); // dim1 hi | tile_dim0=32
      g1[4] = COLS_PI;                               // tile_dim1=128 (tile_dim2=0 -> 2D)
      g1[5] = E;                                     // tensor_dim0_stride[31:0]
      g1[6] = (int)((unsigned)(E & 0xFFFF) << 16);   // dim0_stride hi=0 | dim1_stride lo
      g1[7] = (E >> 16) & 0xFFFF;                    // dim1_stride hi
      i4v g2 = {}; i4v g3 = {};
#if __clang_major__ >= 23
      i8v g4 = {};
      __builtin_amdgcn_tensor_load_to_lds(g0, g1, g2, g3, g4, 0);
#else
      __builtin_amdgcn_tensor_load_to_lds(g0, g1, g2, g3, 0);
#endif
    }
#elif USE_ASM_TR16
    // async global->LDS copy; ASYNCcnt is used by nothing else in this kernel
    const int rowS  = tid & (COLS_PI - 1);
    const int halfS = tid >> 7;
    unsigned long long ga =
        (unsigned long long)(size_t)(p16 + (size_t)(jj + rowS) * E + kk + halfS * 16);
    unsigned la = (unsigned)(size_t)&Bt[bufi][rowS][halfS * 16];
    asm volatile("global_load_async_to_lds_b128 %0, %1, off\n\t"
                 "global_load_async_to_lds_b128 %0, %1, off offset:16"
                 :: "v"(la), "v"(ga) : "memory");
#else
    const int cst   = tid & (COLS_PI - 1);
    const int khalf = tid >> 7;
    const unsigned short* src = p16 + (size_t)(jj + cst) * E + kk + khalf * 16;
    union { f4 q[2]; unsigned short s[16]; } u;
    u.q[0] = ((const f4*)src)[0];
    u.q[1] = ((const f4*)src)[1];
    if (kk + KC < E) __builtin_prefetch(src + KC, 0, 1);
#pragma unroll
    for (int k = 0; k < 16; ++k) Bt[bufi][khalf * 16 + k][cst] = u.s[k];
#endif
  };

  auto pipe_wait = [&]() {   // drain staging before the barrier releases readers
#if USE_ASM_TR16 && HAVE_TDM
    if (wave == 0) __builtin_amdgcn_s_wait_tensorcnt(0);
#elif USE_ASM_TR16
    asm volatile("s_wait_asynccnt 0x0" ::: "memory");
#endif
  };

#if USE_ASM_TR16
  // issue the two direct reads (8 bf16 of row n at k and k+16) for one B fragment
  auto tr16_issue = [&](unsigned rb, i4v& t0, i4v& t1) {
    asm volatile("ds_load_b128 %0, %2\n\t"
                 "ds_load_b128 %1, %2 offset:32"
                 : "=&v"(t0), "=&v"(t1) : "v"(rb));
  };
#endif

  for (int j0 = cbeg; j0 < cbeg + cps; j0 += COLS_PI) {
    v8f acc[8] = {};
    stage(0, j0, 0);
    pipe_wait();
    __syncthreads();
    for (int kc = 0; kc < NK; ++kc) {
      const int buf = kc & 1;
      if (kc + 1 < NK) stage(buf ^ 1, j0, (kc + 1) * KC);  // overlap next chunk

      // A fragment per ISA 16-bit 16x32 layout: lane -> row M, two 16B chunks
      union { v16bf v; f4 q[2]; } Af;
      {
        const f4* ab = (const f4*)(arow + kc * KC + hsel * 8);
        Af.q[0] = ab[0];
        Af.q[1] = ab[2];                        // +32 bytes
      }
#if USE_ASM_TR16
      // software-pipelined direct reads: frag s+1 in flight during WMMA s.
      // All DS ops in this loop are ours -> exact manual DScnt thresholds.
      i4v t0a, t1a, t0b, t1b;
      tr16_issue((unsigned)(size_t)&Bt[buf][lane15][hsel * 8], t0a, t1a);
#pragma unroll
      for (int s = 0; s < 8; ++s) {
        const bool even = (s & 1) == 0;
        if (s + 1 < 8)
          tr16_issue((unsigned)(size_t)&Bt[buf][(s + 1) * 16 + lane15][hsel * 8],
                     even ? t0b : t0a, even ? t1b : t1a);
        i4v& c0 = even ? t0a : t0b;
        i4v& c1 = even ? t1a : t1b;
        if (s + 1 < 8) asm volatile("s_wait_dscnt 0x2" : "+v"(c0), "+v"(c1));
        else           asm volatile("s_wait_dscnt 0x0" : "+v"(c0), "+v"(c1));
        union { v16bf v; i4v q[2]; } Bf;
        Bf.q[0] = c0; Bf.q[1] = c1;
        acc[s] = __builtin_amdgcn_wmma_f32_16x16x32_bf16(
            false, Af.v, false, Bf.v, (short)0, acc[s], false, false);
      }
#else
      // double-buffered B fragments: load frag s+1 while WMMA s executes
      union BFu { v16bf v; f4 q[2]; } Ba, Bb;
      {
        const f4* bb = (const f4*)&Bt[buf][lane][0];
        Ba.q[0] = bb[0]; Ba.q[1] = bb[1];
      }
#pragma unroll
      for (int s = 0; s < 8; ++s) {
        if (s + 1 < 8) {
          const f4* bb = (const f4*)&Bt[buf][lane][(s + 1) * 16];
          if ((s & 1) == 0) { Bb.q[0] = bb[0]; Bb.q[1] = bb[1]; }
          else              { Ba.q[0] = bb[0]; Ba.q[1] = bb[1]; }
        }
        const v16bf bv = ((s & 1) == 0) ? Ba.v : Bb.v;
        acc[s] = __builtin_amdgcn_wmma_f32_16x16x32_bf16(
            false, Af.v, false, bv, (short)0, acc[s], false, false);
      }
#endif
      pipe_wait();
      __syncthreads();
    }
    // epilogue: fold 16x128 sim tile into per-lane running argmax of |sim-1|
#pragma unroll
    for (int s = 0; s < 8; ++s) {
      const int ncol = j0 + s * 16 + lane15;
#pragma unroll
      for (int r = 0; r < 8; ++r) {
        const int mrow = rbase + r + hsel * 8;   // C/D layout: lane half -> M+8
        float sim = acc[s][r];
        float dd  = (ncol == mrow) ? -INFINITY : fabsf(sim - MARGIN);
        dd = (dd == 0.0f) ? -INFINITY : dd;      // reference: d==0 -> -inf
        if (dd > bestD[r]) { bestD[r] = dd; bestS[r] = sim; }
      }
    }
  }

  // cross-lane argmax within each 16-lane half (each half owns distinct rows)
#pragma unroll
  for (int r = 0; r < 8; ++r) {
    float d = bestD[r], s = bestS[r];
#pragma unroll
    for (int m = 1; m < 16; m <<= 1) {
      float d2 = __shfl_xor(d, m, 32);
      float s2 = __shfl_xor(s, m, 32);
      if (d2 > d) { d = d2; s = s2; }
    }
    if (lane15 == 0) {
      const int row = rbase + r + hsel * 8;
      unsigned long long key =
          ((unsigned long long)orderF(d) << 32) | (unsigned long long)orderF(s);
      atomicMax(&packed[row], key);
    }
  }
}

// ---------------------------------------------------------------------------
// Kernel 4: loss = mean(relu(MARGIN + ap - an))
// ---------------------------------------------------------------------------
__global__ __launch_bounds__(256)
void finalize_k(const float* __restrict__ ap,
                const unsigned long long* __restrict__ packed,
                float* __restrict__ out, int B) {
  __shared__ float red[256];
  const int tid = threadIdx.x;
  float s = 0.0f;
  for (int i = tid; i < B; i += 256) {
    float an = unorderF((unsigned)(packed[i] & 0xFFFFFFFFull));
    float l  = MARGIN + ap[i] - an;
    s += (l > 0.0f) ? l : 0.0f;
  }
  red[tid] = s;
  __syncthreads();
  for (int off = 128; off > 0; off >>= 1) {
    if (tid < off) red[tid] += red[tid + off];
    __syncthreads();
  }
  if (tid == 0) out[0] = red[0] / (float)B;
}

// ---------------------------------------------------------------------------
extern "C" void kernel_launch(void* const* d_in, const int* in_sizes, int n_in,
                              void* d_out, int out_size, void* d_ws, size_t ws_size,
                              hipStream_t stream) {
  (void)n_in; (void)out_size; (void)ws_size;
  const float* anchor   = (const float*)d_in[0];
  const float* positive = (const float*)d_in[1];
  const int B = in_sizes[0] / E;                // 8192

  char* w = (char*)d_ws;
  unsigned short* a16 = (unsigned short*)w;                       // B*E bf16
  unsigned short* p16 = a16 + (size_t)B * E;                      // B*E bf16
  float*          ap  = (float*)(p16 + (size_t)B * E);            // B f32
  unsigned long long* packed = (unsigned long long*)(ap + B);     // B u64

  normalize_k<<<B, 256, 0, stream>>>(anchor, positive, a16, p16, ap);
  init_k<<<(B + 255) / 256, 256, 0, stream>>>(packed, B);
  dim3 grid(B / ROWS_PB, NSPLIT);
  mine_k<<<grid, 256, 0, stream>>>(a16, p16, packed, B);
  finalize_k<<<1, 256, 0, stream>>>(ap, packed, (float*)d_out, B);
}
